// SequenceModel_30425548325019
// MI455X (gfx1250) — matrix operand order repaired and checked by reading, not compiled, not run on hardware
//
#include <hip/hip_runtime.h>

// ---------------------------------------------------------------------------
// Types for CDNA5 WMMA (wave32, 16x16x32 bf16 -> f32) and TDM descriptors
// ---------------------------------------------------------------------------
using bf16_t = __bf16;
typedef __attribute__((ext_vector_type(16))) __bf16 v16bf;
typedef __attribute__((ext_vector_type(8)))  float  v8f;
typedef __attribute__((ext_vector_type(4)))  unsigned int v4u;
typedef __attribute__((ext_vector_type(8)))  int v8i;
typedef __attribute__((ext_vector_type(4)))  int v4i;

#if __has_builtin(__builtin_amdgcn_tensor_load_to_lds) && \
    __has_builtin(__builtin_amdgcn_s_wait_tensorcnt)
#define USE_TDM 1
#else
#define USE_TDM 0
#endif

// Model constants
constexpr int kDP = 128, kDT = 256, kDF = 384;
constexpr int kD = 768;               // D
constexpr int kDff = 2048;
constexpr int kNHead = 12;
constexpr int kHd = 64;
constexpr int kCtx = 1024;
constexpr int kPred = 64;
constexpr int kS = 1088;              // CTX + PRED
constexpr int kB = 8;
constexpr int kLayers = 6;
constexpr int kNumPart = 64, kNumTimes = 601, kNumF0 = 360;
constexpr size_t kMR = (size_t)kB * kS;   // 8704 rows
constexpr float kNeg = -1.0e9f;

// ---------------------------------------------------------------------------
// WMMA fragment loaders (wave32; A and B share a contiguous k order, not the ISA table order)
//   A 16x32 bf16: lane L -> row M=L&15; 16 contiguous K at offset (L>>4)*16
//
//   B 32x16 bf16: lane L -> col N=L&15; 16 contiguous K at offset (L>>4)*16
//   C/D f32 16x16: elem r -> row M = r + 8*(L>>4), col N = L&15
// ---------------------------------------------------------------------------
static __device__ __forceinline__ v16bf load_a_frag(const bf16_t* base, int ld) {
  int lane = threadIdx.x & 31;
  int m = lane & 15, half = lane >> 4;
  const bf16_t* p = base + (size_t)m * ld;
  v16bf f;
#pragma unroll
  for (int i = 0; i < 8; ++i) f[i] = p[half * 16 + i];
#pragma unroll
  for (int i = 0; i < 8; ++i) f[8 + i] = p[half * 16 + 8 + i];
  return f;
}

static __device__ __forceinline__ v16bf load_b_frag(const bf16_t* base, int ld) {
  int lane = threadIdx.x & 31;
  int n = lane & 15, half = lane >> 4;
  const bf16_t* p = base + (size_t)n * ld + half * 16;
  v16bf f;
#pragma unroll
  for (int i = 0; i < 16; ++i) f[i] = p[i];
  return f;
}

#if USE_TDM
// ---------------------------------------------------------------------------
// Issue one TDM 2D tile load: tile_rows x 64 bf16 elements, row-major source
// with row stride `stride_elts`; rows >= tensor_rows are zero-filled (used for
// the ragged logit heads). LDS destination gets 16B padding after each 128B
// row (pad_interval code 4 = 32 DWORDs, pad_amount code 3 = 4 DWORDs) ->
// 144B row pitch == LDT(72) bf16, matching the fragment loaders.
// D# field layout per CDNA5 ISA 8.3/8.4. This toolchain exposes the 6-arg
// builtin: (v4u g0, v8i g1, v4i g2, v4i g3, v8i g4, i32 cpol); groups 2/3/4
// are unused for a 2D tensor -> zero-filled.
// ---------------------------------------------------------------------------
static __device__ __forceinline__ void tdm_load_2d(
    const bf16_t* gsrc, unsigned lds_off, unsigned tensor_rows,
    unsigned stride_elts, unsigned tile_rows) {
  unsigned long long ga = (unsigned long long)(size_t)gsrc;
  v4u g0;
  g0[0] = 1u;                                       // count=1, user mode
  g0[1] = lds_off;                                  // lds_addr (bytes)
  g0[2] = (unsigned)(ga & 0xffffffffu);             // global_addr[31:0]
  g0[3] = (unsigned)((ga >> 32) & 0x01ffffffu) | (2u << 30);  // addr hi | type=2
  v8i g1;
  // wg_mask=0 | data_size=1(2B) | pad_enable | pad_interval=4 | pad_amount=3
  g1[0] = (int)((1u << 16) | (1u << 20) | (4u << 22) | (3u << 25));
  unsigned td0 = 64u;                               // tensor_dim0 (elements)
  g1[1] = (int)((td0 & 0xffffu) << 16);             // tensor_dim0[15:0]
  g1[2] = (int)(((td0 >> 16) & 0xffffu) | ((tensor_rows & 0xffffu) << 16));
  g1[3] = (int)(((tensor_rows >> 16) & 0xffffu) | (64u << 16));  // tile_dim0=64
  g1[4] = (int)(tile_rows & 0xffffu);               // tile_dim1, tile_dim2=0
  g1[5] = (int)stride_elts;                         // tensor_dim0_stride[31:0]
  g1[6] = 0;                                        // stride hi, dim1_stride lo
  g1[7] = 0;
  v4i z4 = {0, 0, 0, 0};
  v8i z8 = {0, 0, 0, 0, 0, 0, 0, 0};
  __builtin_amdgcn_tensor_load_to_lds(g0, g1, z4, z4, z8, 0);
}
#endif

// ---------------------------------------------------------------------------
// fp32 -> bf16 conversion (weights, tables)
// ---------------------------------------------------------------------------
__global__ void f2bf(const float* __restrict__ s, bf16_t* __restrict__ d, size_t n) {
  size_t i = (size_t)blockIdx.x * blockDim.x + threadIdx.x;
  size_t st = (size_t)gridDim.x * blockDim.x;
  for (; i < n; i += st) d[i] = (bf16_t)s[i];
}

// ---------------------------------------------------------------------------
// Embedding: [part gather | sin_emb(time) | f0 gather] for ctx rows, pad for
// pred rows. Writes fp32 residual stream and bf16 GEMM operand.
// ---------------------------------------------------------------------------
__global__ __launch_bounds__(256) void embed_kernel(
    const int* __restrict__ cp, const float* __restrict__ ct,
    const int* __restrict__ cf0, const float* __restrict__ part,
    const float* __restrict__ f0t, const float* __restrict__ padv,
    float* __restrict__ x, bf16_t* __restrict__ xb) {
  int r = blockIdx.x;                 // 0..B*S-1
  int b = r / kS, sp = r % kS;
  for (int c = threadIdx.x; c < kD; c += 256) {
    float v;
    if (sp >= kCtx) {
      v = padv[c];
    } else if (c < kDP) {
      v = part[(size_t)cp[b * kCtx + sp] * kDP + c];
    } else if (c < kDP + kDT) {
      int j = (c - kDP) >> 1;
      float wk = __powf(10000.0f, -(float)j / (float)kDT) * ((float)kDT / (float)kNumTimes);
      float ang = ct[b * kCtx + sp] * wk;
      v = ((c - kDP) & 1) ? __cosf(ang) : __sinf(ang);
    } else {
      v = f0t[(size_t)cf0[b * kCtx + sp] * kDF + (c - kDP - kDT)];
    }
    x[(size_t)r * kD + c] = v;
    xb[(size_t)r * kD + c] = (bf16_t)v;
  }
}

// ---------------------------------------------------------------------------
// Generic GEMM: C[M,N] = A[M,K] * W[N,K]^T + bias, optional ReLU.
// Block tile 64x128, 8 waves x (32x32), K step 64 (8 WMMA / iteration).
// TDM path: double-buffered LDS tiles; wave 0 issues two tensor_load_to_lds
// per step (A 64x64, W 128x64) and s_wait_tensorcnt gates the barrier, so the
// DMA of tile t+1 overlaps WMMA on tile t.
// ---------------------------------------------------------------------------
__global__ __launch_bounds__(256) void gemm_bf16(
    const bf16_t* __restrict__ A, int lda,
    const bf16_t* __restrict__ W, int ldw,
    const float* __restrict__ bias,
    float* __restrict__ Cf, bf16_t* __restrict__ Cb, int ldc,
    int M, int N, int K, int relu) {
  constexpr int LDT = 72;             // 144B pitch: aligned + conflict-free
#if USE_TDM
  __shared__ bf16_t As[2][64][LDT];
  __shared__ bf16_t Bs[2][128][LDT];
#else
  __shared__ bf16_t As[1][64][LDT];
  __shared__ bf16_t Bs[1][128][LDT];
#endif

  int bm = blockIdx.y * 64;
  int bn = blockIdx.x * 128;
  int tid = threadIdx.x, lane = tid & 31, wv = tid >> 5;
  int wm = (wv >> 2) * 32;            // 0,32
  int wn = (wv & 3) * 32;             // 0..96
  int halfc = lane >> 4, nc = lane & 15;
  v8f acc[2][2] = {};
  int nk = K >> 6;

#if USE_TDM
  if (tid < 32) {                     // wave 0 issues the prologue DMAs
    tdm_load_2d(A + (size_t)bm * lda, (unsigned)(size_t)&As[0][0][0],
                64u, (unsigned)lda, 64u);
    tdm_load_2d(W + (size_t)bn * ldw, (unsigned)(size_t)&Bs[0][0][0],
                (unsigned)(N - bn), (unsigned)ldw, 128u);
  }
#endif

  for (int t = 0; t < nk; ++t) {
#if USE_TDM
    int cur = t & 1;
    if (tid < 32) __builtin_amdgcn_s_wait_tensorcnt(0);
    __syncthreads();                  // buf[cur] ready; buf[cur^1] free
    if (tid < 32 && t + 1 < nk) {
      int k0n = (t + 1) << 6;
      tdm_load_2d(A + (size_t)bm * lda + k0n,
                  (unsigned)(size_t)&As[cur ^ 1][0][0], 64u, (unsigned)lda, 64u);
      tdm_load_2d(W + (size_t)bn * ldw + k0n,
                  (unsigned)(size_t)&Bs[cur ^ 1][0][0],
                  (unsigned)(N - bn), (unsigned)ldw, 128u);
    }
#else
    constexpr int cur = 0;
    int k0 = t << 6;
    __syncthreads();
    {
      int rowa = tid >> 2, sega = (tid & 3) * 16;
      const bf16_t* src = A + (size_t)(bm + rowa) * lda + k0 + sega;
#pragma unroll
      for (int i = 0; i < 16; ++i) As[0][rowa][sega + i] = src[i];
    }
    {
      int rowb = tid >> 1, segb = (tid & 1) * 32;
      int gn = bn + rowb;
      if (gn < N) {
        const bf16_t* src = W + (size_t)gn * ldw + k0 + segb;
#pragma unroll
        for (int i = 0; i < 32; ++i) Bs[0][rowb][segb + i] = src[i];
      } else {
#pragma unroll
        for (int i = 0; i < 32; ++i) Bs[0][rowb][segb + i] = (bf16_t)0.0f;
      }
    }
    __syncthreads();
#endif
#pragma unroll
    for (int kk = 0; kk < 2; ++kk) {
      v16bf a0 = load_a_frag(&As[cur][wm][kk * 32], LDT);
      v16bf a1 = load_a_frag(&As[cur][wm + 16][kk * 32], LDT);
#pragma unroll
      for (int j = 0; j < 2; ++j) {
        v16bf bfr = load_b_frag(&Bs[cur][wn + j * 16][kk * 32], LDT);
        acc[0][j] = __builtin_amdgcn_wmma_f32_16x16x32_bf16(
            false, a0, false, bfr, (short)0, acc[0][j], false, false);
        acc[1][j] = __builtin_amdgcn_wmma_f32_16x16x32_bf16(
            false, a1, false, bfr, (short)0, acc[1][j], false, false);
      }
    }
  }

#pragma unroll
  for (int i = 0; i < 2; ++i) {
#pragma unroll
    for (int j = 0; j < 2; ++j) {
      int col = bn + wn + j * 16 + nc;
      if (col >= N) continue;
      float bv = bias ? bias[col] : 0.0f;
#pragma unroll
      for (int r = 0; r < 8; ++r) {
        int rowg = bm + wm + i * 16 + r + 8 * halfc;
        float v = acc[i][j][r] + bv;
        if (relu) v = fmaxf(v, 0.0f);
        size_t idx = (size_t)rowg * ldc + col;
        if (Cf) Cf[idx] = v;
        if (Cb) Cb[idx] = (bf16_t)v;
      }
    }
  }
  (void)M;
}

// ---------------------------------------------------------------------------
// Attention: flash-style, one workgroup per (64-row q tile, head, batch).
// Q frags in registers (scale folded), K frags direct from global (B layout),
// scores -> LDS -> online softmax -> P(bf16, LDS) x V^T(LDS) WMMA.
// Causal mask + pred-pred block mask applied in registers.
// ---------------------------------------------------------------------------
__global__ __launch_bounds__(256) void attn_kernel(
    const bf16_t* __restrict__ qkv, bf16_t* __restrict__ outp) {
  constexpr int LDT = 72;             // 144B stride: aligned, conflict-free
  __shared__ bf16_t Vt[64][LDT];      // V transposed: [hd][kpos]
  __shared__ bf16_t Ps[64][LDT];      // exp probabilities (A layout source)
  __shared__ float  Sc[64][LDT];      // raw scores
  __shared__ float  alpha_s[64];
  __shared__ float  linv_s[64];

  int qt = blockIdx.x, h = blockIdx.y, b = blockIdx.z;
  int q0 = qt * 64;
  int tid = threadIdx.x, lane = tid & 31, wv = tid >> 5;
  int sm = (wv >> 1) * 16;            // 0,16,32,48
  int sn = (wv & 1) * 32;             // 0,32
  int halfc = lane >> 4, nc = lane & 15;
  const size_t ldq = 3 * kD;
  const bf16_t* base = qkv + (size_t)b * kS * ldq;
  const float scale = 0.125f;         // 1/sqrt(64)

  // Q fragments (scaled), for K-chunks 0..31 and 32..63 of the head dim
  v16bf qf[2];
  {
    int m = lane & 15;
    const bf16_t* qrow = base + (size_t)(q0 + sm + m) * ldq + h * kHd;
#pragma unroll
    for (int kk = 0; kk < 2; ++kk) {
#pragma unroll
      for (int i = 0; i < 8; ++i)
        qf[kk][i] = (bf16_t)((float)qrow[kk * 32 + halfc * 16 + i] * scale);
#pragma unroll
      for (int i = 0; i < 8; ++i)
        qf[kk][8 + i] = (bf16_t)((float)qrow[kk * 32 + halfc * 16 + 8 + i] * scale);
    }
  }

  v8f oacc[2] = {};
  float m_run = -3.0e38f, l_run = 0.0f;

  int ktend = q0 + 64;                // causal upper bound for this q tile
  for (int kt0 = 0; kt0 < ktend; kt0 += 64) {
    __syncthreads();
    // stage V transposed into LDS
    {
      int kp = tid >> 2;
      int cs = (tid & 3) * 16;
      const bf16_t* vrow = base + (size_t)(kt0 + kp) * ldq + 2 * kD + h * kHd + cs;
#pragma unroll
      for (int i = 0; i < 16; ++i) Vt[cs + i][kp] = vrow[i];
    }
    // scores = Q * K^T  (K frags straight from global in B layout)
    v8f sacc[2] = {};
#pragma unroll
    for (int kk = 0; kk < 2; ++kk) {
#pragma unroll
      for (int j = 0; j < 2; ++j) {
        v16bf kf;
        {
          int n = lane & 15;
          const bf16_t* krow = base + (size_t)(kt0 + sn + j * 16 + n) * ldq +
                               kD + h * kHd + kk * 32 + halfc * 16;
#pragma unroll
          for (int i = 0; i < 16; ++i) kf[i] = krow[i];
        }
        sacc[j] = __builtin_amdgcn_wmma_f32_16x16x32_bf16(
            false, qf[kk], false, kf, (short)0, sacc[j], false, false);
      }
    }
    // mask + spill to LDS
#pragma unroll
    for (int j = 0; j < 2; ++j) {
      int ncol = sn + j * 16 + nc;
      int kg = kt0 + ncol;
#pragma unroll
      for (int r = 0; r < 8; ++r) {
        int mrow = sm + r + 8 * halfc;
        int qg = q0 + mrow;
        float v = sacc[j][r];
        if (kg > qg || (qg >= kCtx && kg >= kCtx)) v = kNeg;
        Sc[mrow][ncol] = v;
      }
    }
    __syncthreads();
    // online softmax (one thread per row)
    if (tid < 64) {
      float mx = -3.0e38f;
#pragma unroll 8
      for (int c = 0; c < 64; ++c) mx = fmaxf(mx, Sc[tid][c]);
      float mnew = fmaxf(m_run, mx);
      float al = __expf(m_run - mnew);
      float s = 0.0f;
#pragma unroll 8
      for (int c = 0; c < 64; ++c) {
        float p = __expf(Sc[tid][c] - mnew);
        Ps[tid][c] = (bf16_t)p;
        s += p;
      }
      l_run = l_run * al + s;
      m_run = mnew;
      alpha_s[tid] = al;
    }
    __syncthreads();
    // rescale accumulators, then O += P * V
#pragma unroll
    for (int j = 0; j < 2; ++j)
#pragma unroll
      for (int r = 0; r < 8; ++r) oacc[j][r] *= alpha_s[sm + r + 8 * halfc];
#pragma unroll
    for (int kk = 0; kk < 2; ++kk) {
      v16bf pf = load_a_frag(&Ps[sm][kk * 32], LDT);
#pragma unroll
      for (int j = 0; j < 2; ++j) {
        v16bf vf = load_b_frag(&Vt[sn + j * 16][kk * 32], LDT);
        oacc[j] = __builtin_amdgcn_wmma_f32_16x16x32_bf16(
            false, pf, false, vf, (short)0, oacc[j], false, false);
      }
    }
  }
  __syncthreads();
  if (tid < 64) linv_s[tid] = 1.0f / l_run;
  __syncthreads();
#pragma unroll
  for (int j = 0; j < 2; ++j) {
    int ncol = sn + j * 16 + nc;
#pragma unroll
    for (int r = 0; r < 8; ++r) {
      int mrow = sm + r + 8 * halfc;
      float v = oacc[j][r] * linv_s[mrow];
      outp[((size_t)b * kS + q0 + mrow) * kD + h * kHd + ncol] = (bf16_t)v;
    }
  }
}

// ---------------------------------------------------------------------------
// Residual + LayerNorm: x = LN(x + proj)*g + b ; writes fp32 + bf16.
// ---------------------------------------------------------------------------
__global__ __launch_bounds__(256) void resid_ln(
    float* __restrict__ x, const float* __restrict__ proj,
    const float* __restrict__ g, const float* __restrict__ bta,
    bf16_t* __restrict__ xb) {
  size_t row = blockIdx.x;
  int t = threadIdx.x;
  float v[3], s1 = 0.0f, s2 = 0.0f;
#pragma unroll
  for (int i = 0; i < 3; ++i) {
    int c = t + i * 256;
    float val = x[row * kD + c] + proj[row * kD + c];
    v[i] = val; s1 += val; s2 += val * val;
  }
#pragma unroll
  for (int off = 16; off > 0; off >>= 1) {
    s1 += __shfl_down(s1, off, 32);
    s2 += __shfl_down(s2, off, 32);
  }
  __shared__ float ws1[8], ws2[8];
  __shared__ float mu_s, rs_s;
  int lane = t & 31, w = t >> 5;
  if (lane == 0) { ws1[w] = s1; ws2[w] = s2; }
  __syncthreads();
  if (t == 0) {
    float a = 0.0f, q = 0.0f;
#pragma unroll
    for (int i = 0; i < 8; ++i) { a += ws1[i]; q += ws2[i]; }
    float mu = a / (float)kD;
    float var = q / (float)kD - mu * mu;
    mu_s = mu; rs_s = rsqrtf(var + 1e-5f);
  }
  __syncthreads();
  float mu = mu_s, rs = rs_s;
#pragma unroll
  for (int i = 0; i < 3; ++i) {
    int c = t + i * 256;
    float y = (v[i] - mu) * rs * g[c] + bta[c];
    x[row * kD + c] = y;
    xb[row * kD + c] = (bf16_t)y;
  }
}

// ---------------------------------------------------------------------------
// Gather the 64 prediction rows per batch into a dense 512x768 bf16 matrix.
// ---------------------------------------------------------------------------
__global__ __launch_bounds__(256) void gather_pred(
    const bf16_t* __restrict__ xb, bf16_t* __restrict__ pred) {
  int r = blockIdx.x;                 // 0..511
  int b = r / kPred, p = r % kPred;
  const bf16_t* src = xb + ((size_t)b * kS + kCtx + p) * kD;
  for (int c = threadIdx.x; c < kD; c += 256) pred[(size_t)r * kD + c] = src[c];
}

// ---------------------------------------------------------------------------
// Host orchestration
// ---------------------------------------------------------------------------
extern "C" void kernel_launch(void* const* d_in, const int* in_sizes, int n_in,
                              void* d_out, int out_size, void* d_ws, size_t ws_size,
                              hipStream_t stream) {
  (void)in_sizes; (void)n_in; (void)out_size; (void)ws_size;
  const int*   cp    = (const int*)  d_in[0];
  const float* ct    = (const float*)d_in[1];
  const int*   cf0   = (const int*)  d_in[2];
  const float* part  = (const float*)d_in[3];
  const float* timet = (const float*)d_in[4];
  const float* f0t   = (const float*)d_in[5];
  const float* padv  = (const float*)d_in[6];
  const float* Wqkv  = (const float*)d_in[7];
  const float* bqkv  = (const float*)d_in[8];
  const float* Wo    = (const float*)d_in[9];
  const float* bo    = (const float*)d_in[10];
  const float* ln1g  = (const float*)d_in[11];
  const float* ln1b  = (const float*)d_in[12];
  const float* W1    = (const float*)d_in[13];
  const float* b1    = (const float*)d_in[14];
  const float* W2    = (const float*)d_in[15];
  const float* b2    = (const float*)d_in[16];
  const float* ln2g  = (const float*)d_in[17];
  const float* ln2b  = (const float*)d_in[18];
  float* outp = (float*)d_out;

  char* ws = (char*)d_ws;
  size_t off = 0;
  auto take = [&](size_t bytes) -> char* {
    char* p = ws + off;
    off += (bytes + 255) & ~(size_t)255;
    return p;
  };
  bf16_t* xb    = (bf16_t*)take(kMR * kD * 2);
  float*  x     = (float*) take(kMR * kD * 4);
  bf16_t* qkvb  = (bf16_t*)take(kMR * 3 * kD * 2);
  bf16_t* attnb = (bf16_t*)take(kMR * kD * 2);
  float*  proj  = (float*) take(kMR * kD * 4);
  bf16_t* ffh   = (bf16_t*)take(kMR * kDff * 2);
  bf16_t* predb = (bf16_t*)take((size_t)kB * kPred * kD * 2);
  bf16_t* wqkvb = (bf16_t*)take((size_t)kLayers * 3 * kD * kD * 2);
  bf16_t* wob   = (bf16_t*)take((size_t)kLayers * kD * kD * 2);
  bf16_t* w1b   = (bf16_t*)take((size_t)kLayers * kDff * kD * 2);
  bf16_t* w2b   = (bf16_t*)take((size_t)kLayers * kD * kDff * 2);
  bf16_t* partb = (bf16_t*)take((size_t)kNumPart * kDP * 2);
  bf16_t* timeb = (bf16_t*)take((size_t)kNumTimes * kDT * 2);
  bf16_t* f0b   = (bf16_t*)take((size_t)kNumF0 * kDF * 2);

  // one-time bf16 conversions
  f2bf<<<2048, 256, 0, stream>>>(Wqkv, wqkvb, (size_t)kLayers * 3 * kD * kD);
  f2bf<<<2048, 256, 0, stream>>>(Wo,   wob,   (size_t)kLayers * kD * kD);
  f2bf<<<2048, 256, 0, stream>>>(W1,   w1b,   (size_t)kLayers * kDff * kD);
  f2bf<<<2048, 256, 0, stream>>>(W2,   w2b,   (size_t)kLayers * kD * kDff);
  f2bf<<<32,   256, 0, stream>>>(part,  partb, (size_t)kNumPart * kDP);
  f2bf<<<256,  256, 0, stream>>>(timet, timeb, (size_t)kNumTimes * kDT);
  f2bf<<<256,  256, 0, stream>>>(f0t,   f0b,   (size_t)kNumF0 * kDF);

  embed_kernel<<<(int)kMR, 256, 0, stream>>>(cp, ct, cf0, part, f0t, padv, x, xb);

  const int gM = (int)(kMR / 64);     // 136
  for (int l = 0; l < kLayers; ++l) {
    // QKV projection: (8704 x 768) x (2304 x 768)^T -> bf16 qkv
    gemm_bf16<<<dim3(3 * kD / 128, gM), 256, 0, stream>>>(
        xb, kD, wqkvb + (size_t)l * 3 * kD * kD, kD, bqkv + (size_t)l * 3 * kD,
        nullptr, qkvb, 3 * kD, (int)kMR, 3 * kD, kD, 0);
    // attention
    attn_kernel<<<dim3(kS / 64, kNHead, kB), 256, 0, stream>>>(qkvb, attnb);
    // output projection -> fp32
    gemm_bf16<<<dim3(kD / 128, gM), 256, 0, stream>>>(
        attnb, kD, wob + (size_t)l * kD * kD, kD, bo + (size_t)l * kD,
        proj, nullptr, kD, (int)kMR, kD, kD, 0);
    resid_ln<<<(int)kMR, 256, 0, stream>>>(x, proj, ln1g + (size_t)l * kD,
                                           ln1b + (size_t)l * kD, xb);
    // FFN up + ReLU -> bf16 hidden
    gemm_bf16<<<dim3(kDff / 128, gM), 256, 0, stream>>>(
        xb, kD, w1b + (size_t)l * kDff * kD, kD, b1 + (size_t)l * kDff,
        nullptr, ffh, kDff, (int)kMR, kDff, kD, 1);
    // FFN down -> fp32
    gemm_bf16<<<dim3(kD / 128, gM), 256, 0, stream>>>(
        ffh, kDff, w2b + (size_t)l * kD * kDff, kDff, b2 + (size_t)l * kD,
        proj, nullptr, kD, (int)kMR, kD, kDff, 0);
    resid_ln<<<(int)kMR, 256, 0, stream>>>(x, proj, ln2g + (size_t)l * kD,
                                           ln2b + (size_t)l * kD, xb);
  }

  gather_pred<<<kB * kPred, 256, 0, stream>>>(xb, predb);

  // logit heads (512 rows each), written straight into d_out
  gemm_bf16<<<dim3(1, kB * kPred / 64), 256, 0, stream>>>(
      predb, kD, partb, kDP, nullptr, outp, nullptr,
      kNumPart, kB * kPred, kNumPart, kDP, 0);
  gemm_bf16<<<dim3((kNumTimes + 127) / 128, kB * kPred / 64), 256, 0, stream>>>(
      predb + kDP, kD, timeb, kDT, nullptr, outp + (size_t)kB * kPred * kNumPart,
      nullptr, kNumTimes, kB * kPred, kNumTimes, kDT, 0);
  gemm_bf16<<<dim3((kNumF0 + 127) / 128, kB * kPred / 64), 256, 0, stream>>>(
      predb + kDP + kDT, kD, f0b, kDF, nullptr,
      outp + (size_t)kB * kPred * (kNumPart + kNumTimes),
      nullptr, kNumF0, kB * kPred, kNumF0, kDF, 0);
}
